// MeanFieldCRF_57329223467055
// MI455X (gfx1250) — compile-verified, matrix operand order checked by reading, not run on hardware
//
#include <hip/hip_runtime.h>
#include <hip/hip_bf16.h>
#include <math.h>

typedef __attribute__((ext_vector_type(16))) _Float16 v16h;
typedef __attribute__((ext_vector_type(8)))  float    v8f;

#define B_   16
#define T_   512
#define HD_  16      // per-direction hidden
#define H_   32      // concat hidden
#define G_   64      // 4*HD gates

__device__ __forceinline__ float sigf(float x) { return 1.0f / (1.0f + expf(-x)); }

__device__ __forceinline__ float wave_reduce_add(float v) {
  #pragma unroll
  for (int m = 16; m >= 1; m >>= 1) v += __shfl_xor(v, m, 32);
  return v;
}

// ---------------------------------------------------------------------------
// Kernel 1: bidirectional LSTM scan. grid = 2 blocks (dir), block = 1024.
// thread t: b = t/64, g = t%64 computes gate g of batch b each step.
// out[t][b][dir*16 + h] written for all t.
// ---------------------------------------------------------------------------
__global__ __launch_bounds__(1024) void lstm_kernel(
    const float* __restrict__ sentence,   // [B,T]
    const float* __restrict__ h0,         // [2,B,HD]
    const float* __restrict__ c0,         // [2,B,HD]
    const float* __restrict__ Wih_f, const float* __restrict__ Whh_f, const float* __restrict__ b_f,
    const float* __restrict__ Wih_b, const float* __restrict__ Whh_b, const float* __restrict__ b_b,
    float* __restrict__ outbuf)           // [T,B,H]
{
  __shared__ float sh_h[B_][HD_];
  __shared__ float sh_c[B_][HD_];
  __shared__ float sh_g[B_][G_];
  __shared__ float sh_x[B_][T_];

  const int dir = blockIdx.x;             // 0 = forward, 1 = backward
  const int tid = threadIdx.x;
  const int b   = tid >> 6;
  const int g   = tid & 63;

  const float* Wih = dir ? Wih_b : Wih_f;
  const float* Whh = dir ? Whh_b : Whh_f;
  const float* bv  = dir ? b_b   : b_f;

  for (int idx = tid; idx < B_ * T_; idx += 1024)
    sh_x[idx >> 9][idx & (T_ - 1)] = sentence[idx];

  if (g < HD_) {
    sh_h[b][g] = h0[dir * B_ * HD_ + b * HD_ + g];
    sh_c[b][g] = c0[dir * B_ * HD_ + b * HD_ + g];
  }

  float wrow[HD_];
  #pragma unroll
  for (int k = 0; k < HD_; ++k) wrow[k] = Whh[g * HD_ + k];
  const float wih  = Wih[g];
  const float bias = bv[g];

  __syncthreads();

  for (int s = 0; s < T_; ++s) {
    const int tpos = dir ? (T_ - 1 - s) : s;
    const float x  = sh_x[b][tpos];
    float acc = fmaf(wih, x, bias);
    #pragma unroll
    for (int k = 0; k < HD_; ++k) acc = fmaf(wrow[k], sh_h[b][k], acc);
    sh_g[b][g] = acc;
    __syncthreads();
    if (g < HD_) {
      const float ig = sh_g[b][g];
      const float fg = sh_g[b][HD_ + g];
      const float gg = sh_g[b][2 * HD_ + g];
      const float og = sh_g[b][3 * HD_ + g];
      const float c  = sigf(fg) * sh_c[b][g] + sigf(ig) * tanhf(gg);
      const float h  = sigf(og) * tanhf(c);
      sh_c[b][g] = c;
      sh_h[b][g] = h;
      outbuf[(tpos * B_ + b) * H_ + dir * HD_ + g] = h;
    }
    __syncthreads();
  }
}

// ---------------------------------------------------------------------------
// Kernel 2: A' = out @ W1[:, :32].T + b1 ; C = out @ W1[:, 32:].T  via WMMA.
// out viewed row-major [M=8192, K=32]; combined N=64 (cols 0..31 -> A with b1
// folded, 32..63 -> C). Rearranged B[k][n] is staged in LDS once per block;
// each wave owns one 16-row tile and issues 4 v_wmma (one per 16-col tile),
// reusing its A fragment. 64 blocks x 8 waves = 512 row tiles.
// ---------------------------------------------------------------------------
__global__ __launch_bounds__(256) void proj_wmma_kernel(
    const float* __restrict__ outbuf,  // [M,32] f32
    const float* __restrict__ W1,      // [32,64]
    const float* __restrict__ b1,      // [32]
    float* __restrict__ Aarr,          // [M,32]
    float* __restrict__ Carr)          // [M,32]
{
  __shared__ float shWB[G_][H_];       // shWB[n][k] = B[k][n]  (8 KB)

  const int tid = threadIdx.x;
  for (int idx = tid; idx < G_ * H_; idx += 256) {
    const int n = idx >> 5;
    const int k = idx & 31;
    shWB[n][k] = (n < H_) ? W1[n * 64 + k] : W1[(n - H_) * 64 + H_ + k];
  }
  __syncthreads();

  const int wave = blockIdx.x * 8 + (tid >> 5);   // 0..511 row tiles
  const int lane = tid & 31;
  const int r    = lane & 15;
  const int half = lane >> 4;
  const int m0   = wave * 16;

  // A fragment: row m0+r, k in {8h..8h+7} U {16+8h..16+8h+7} as two float4 pairs
  const float4* rowp = (const float4*)(outbuf + (m0 + r) * H_);
  const float4 q0 = rowp[half * 2 + 0];
  const float4 q1 = rowp[half * 2 + 1];
  const float4 q2 = rowp[4 + half * 2 + 0];
  const float4 q3 = rowp[4 + half * 2 + 1];
  v16h afrag;
  afrag[0]  = (_Float16)q0.x; afrag[1]  = (_Float16)q0.y;
  afrag[2]  = (_Float16)q0.z; afrag[3]  = (_Float16)q0.w;
  afrag[4]  = (_Float16)q1.x; afrag[5]  = (_Float16)q1.y;
  afrag[6]  = (_Float16)q1.z; afrag[7]  = (_Float16)q1.w;
  afrag[8]  = (_Float16)q2.x; afrag[9]  = (_Float16)q2.y;
  afrag[10] = (_Float16)q2.z; afrag[11] = (_Float16)q2.w;
  afrag[12] = (_Float16)q3.x; afrag[13] = (_Float16)q3.y;
  afrag[14] = (_Float16)q3.z; afrag[15] = (_Float16)q3.w;

  #pragma unroll
  for (int ct = 0; ct < 4; ++ct) {
    const int n = ct * 16 + r;                 // this lane's output column
    const float* wb = &shWB[n][0];
    v16h bfrag;
    #pragma unroll
    for (int e = 0; e < 16; ++e) {
      const int k = e + (e >= 8 ? 8 : 0) + half * 8;
      bfrag[e] = (_Float16)wb[k];
    }

    v8f cacc = {};
    cacc = __builtin_amdgcn_wmma_f32_16x16x32_f16(
        /*neg_a=*/false, afrag, /*neg_b=*/false, bfrag,
        /*c_mod=*/(short)0, cacc, /*reuse_a=*/false, /*reuse_b=*/false);

    // lane-constant destination base & bias: no divergence in the store loop
    const float bias = (n < H_) ? b1[n] : 0.0f;
    float* dst = (n < H_) ? (Aarr + n) : (Carr + (n - H_));
    #pragma unroll
    for (int v = 0; v < 8; ++v) {
      const int row = m0 + v + half * 8;       // D: VGPR v -> row v (+8 hi half)
      dst[row * H_] = cacc[v] + bias;
    }
  }
}

// ---------------------------------------------------------------------------
// Kernel 3: block = (b, tile of 8 i's). Whole C[.,b,:] (64 KB) staged in LDS
// once, then 8 waves stream it with conflict-free ds loads. Per (b,i) wave:
// S = sum_{j!=i} sum_h relu(A'[i]+C[j])*W2[h] + 511*b2 ; unary fused;
// prob[b][i] = sigmoid(S/100 + unary).  Grid 1024 blocks x 256 threads.
// ---------------------------------------------------------------------------
__global__ __launch_bounds__(256) void binary_prob_kernel(
    const float* __restrict__ Aarr,    // [T*B,32] (b1 folded)
    const float* __restrict__ Carr,    // [T*B,32]
    const float* __restrict__ outbuf,  // [T*B,32]
    const float* __restrict__ W2,      // [32]
    const float* __restrict__ b2p,     // [1]
    const float* __restrict__ Wu,      // [32]
    const float* __restrict__ bup,     // [1]
    float* __restrict__ probArr)       // [B,T]
{
  __shared__ float shC[T_][H_];        // 64 KB: all of C for this batch

  const int tid   = threadIdx.x;
  const int b     = blockIdx.x & (B_ - 1);
  const int itile = blockIdx.x >> 4;   // 0..63

  for (int idx = tid; idx < T_ * H_; idx += 256) {
    const int j = idx >> 5;
    const int h = idx & 31;
    shC[j][h] = Carr[(j * B_ + b) * H_ + h];
  }
  __syncthreads();

  const int i = itile * 8 + (tid >> 5);
  const int h = tid & 31;

  const int base_i = (i * B_ + b) * H_;
  const float ai  = Aarr[base_i + h];
  const float w2h = W2[h];
  const float up  = outbuf[base_i + h] * Wu[h];   // unary partial

  float acc = 0.0f;
  #pragma unroll 8
  for (int j = 0; j < T_; ++j)
    acc = fmaf(fmaxf(ai + shC[j][h], 0.0f), w2h, acc);

  const float diag = fmaxf(ai + shC[i][h], 0.0f) * w2h;

  const float S = wave_reduce_add(acc - diag);
  const float u = wave_reduce_add(up);

  if (h == 0) {
    const float b2 = b2p[0], bu = bup[0];
    const float logit = (S + (float)(T_ - 1) * b2) * 0.01f + u + bu;
    probArr[b * T_ + i] = sigf(logit);
  }
}

// ---------------------------------------------------------------------------
// Kernel 4: weighted pooling + output linear. 1 block, 16 waves (wave = b).
// ---------------------------------------------------------------------------
__global__ __launch_bounds__(512) void pool_out_kernel(
    const float* __restrict__ probArr,  // [B,T]
    const float* __restrict__ outbuf,   // [T*B,32]
    const float* __restrict__ Wout,     // [32]
    const float* __restrict__ boutp,    // [1]
    float* __restrict__ d_out)          // [B]
{
  const int b = threadIdx.x >> 5;
  const int h = threadIdx.x & 31;

  float acc = 0.0f;
  for (int t = 0; t < T_; ++t)
    acc = fmaf(probArr[b * T_ + t], outbuf[(t * B_ + b) * H_ + h], acc);

  const float red = wave_reduce_add(acc * Wout[h]);
  if (h == 0) d_out[b] = red * (1.0f / (float)T_) + boutp[0];
}

// ---------------------------------------------------------------------------
extern "C" void kernel_launch(void* const* d_in, const int* in_sizes, int n_in,
                              void* d_out, int out_size, void* d_ws, size_t ws_size,
                              hipStream_t stream) {
  const float* sentence = (const float*)d_in[0];
  const float* h0       = (const float*)d_in[1];
  const float* c0       = (const float*)d_in[2];
  const float* Wih_f    = (const float*)d_in[3];
  const float* Whh_f    = (const float*)d_in[4];
  const float* b_f      = (const float*)d_in[5];
  const float* Wih_b    = (const float*)d_in[6];
  const float* Whh_b    = (const float*)d_in[7];
  const float* b_b      = (const float*)d_in[8];
  const float* W1       = (const float*)d_in[9];
  const float* b1       = (const float*)d_in[10];
  const float* W2       = (const float*)d_in[11];
  const float* b2       = (const float*)d_in[12];
  const float* Wu       = (const float*)d_in[13];
  const float* bu       = (const float*)d_in[14];
  const float* Wout     = (const float*)d_in[15];
  const float* bout     = (const float*)d_in[16];

  float* ws = (float*)d_ws;
  const size_t M = (size_t)T_ * B_;          // 8192
  float* outbuf  = ws;                        // M*32 floats
  float* Aarr    = outbuf + M * H_;           // M*32
  float* Carr    = Aarr   + M * H_;           // M*32
  float* probArr = Carr   + M * H_;           // B*T

  lstm_kernel<<<2, 1024, 0, stream>>>(sentence, h0, c0,
                                      Wih_f, Whh_f, b_f,
                                      Wih_b, Whh_b, b_b, outbuf);

  proj_wmma_kernel<<<64, 256, 0, stream>>>(outbuf, W1, b1, Aarr, Carr);

  binary_prob_kernel<<<1024, 256, 0, stream>>>(Aarr, Carr, outbuf,
                                               W2, b2, Wu, bu, probArr);

  pool_out_kernel<<<1, 512, 0, stream>>>(probArr, outbuf, Wout, bout,
                                         (float*)d_out);
}
